// LSTMCaptioner_51436528337134
// MI455X (gfx1250) — matrix operand order repaired and checked by reading, not compiled, not run on hardware
//
#include <hip/hip_runtime.h>
#include <hip/hip_bf16.h>
#include <math.h>

typedef __attribute__((ext_vector_type(16))) _Float16 v16h;
typedef __attribute__((ext_vector_type(8)))  _Float16 v8h;
typedef __attribute__((ext_vector_type(8)))  float    v8f;

// Problem constants
#define BB 8
#define NQ 32
#define SS 21
#define DD 256
#define HH 8
#define VV 10000
#define HD 32
#define TT 1920
#define NE 256            // B*NQ
#define STEPS 20          // S-1

// ---------------------------------------------------------------------------
// WMMA fragment helpers (wave32, gfx1250 layouts)
// A (16x32 f16): lane holds row lane&15; 16 halves = contiguous run
//   at k = kk + 16*(lane>>4) .. +15  -> two b128 loads + shuffle.
// B (32x16 f16): pre-packed so each lane's 16 halves are contiguous (32B load).
// C/D (16x16 f32): reg r -> row r + 8*(lane>>4), col lane&15.
// ---------------------------------------------------------------------------
__device__ __forceinline__ v16h load_a_rm(const _Float16* __restrict__ p, int hi) {
  const v8h* q = (const v8h*)(p + (hi << 4));
  v8h a0 = q[0];   // k+0..7
  v8h a1 = q[1];   // k+8..15
  return __builtin_shufflevector(a0, a1, 0, 1, 2, 3, 4, 5, 6, 7, 8, 9, 10, 11, 12, 13, 14, 15);
}
__device__ __forceinline__ v16h load_b_pk(const _Float16* __restrict__ packed, int frag, int lane) {
  return *(const v16h*)(packed + (((size_t)frag * 32 + lane) << 4));
}
#define WMMA(a, b, c) __builtin_amdgcn_wmma_f32_16x16x32_f16(false, (a), false, (b), (short)0, (c), false, false)

// ---------------------------------------------------------------------------
// setup / packing kernels
// ---------------------------------------------------------------------------
__global__ void pack16(const float* __restrict__ src, _Float16* __restrict__ dst, int n) {
  int i = blockIdx.x * 256 + threadIdx.x;
  if (i < n) dst[i] = (_Float16)src[i];
}

// generic: src row-major K x N f32 -> WMMA-B fragment-packed f16
__global__ void packB_f32(const float* __restrict__ src, _Float16* __restrict__ dst, int K, int N) {
  int i = blockIdx.x * 256 + threadIdx.x;
  if (i >= K * N) return;
  int e = i & 15, lane = (i >> 4) & 31;
  int rest = i >> 9;                 // nt*KT + kt
  int KT = K >> 5;
  int kt = rest % KT, nt = rest / KT;
  int k = kt * 32 + e + ((lane >> 4) << 4);
  int n = nt * 16 + (lane & 15);
  dst[i] = (_Float16)src[(size_t)k * N + n];
}

// Wcomb (K=512,N=512): cols [0,128)=off_w, [128,256)=aw_w, [256,512)=hs_w(rows<256) else 0
__global__ void packB_wcomb(const float* __restrict__ off_w, const float* __restrict__ aw_w,
                            const float* __restrict__ hs_w, _Float16* __restrict__ dst) {
  int i = blockIdx.x * 256 + threadIdx.x;
  if (i >= 512 * 512) return;
  int e = i & 15, lane = (i >> 4) & 31;
  int rest = i >> 9;
  int kt = rest & 15, nt = rest >> 4;      // KT=16
  int k = kt * 32 + e + ((lane >> 4) << 4);
  int j = nt * 16 + (lane & 15);
  float v;
  if (j < 128)      v = off_w[k * 128 + j];
  else if (j < 256) v = aw_w[k * 128 + (j - 128)];
  else              v = (k < 256) ? hs_w[k * 256 + (j - 256)] : 0.0f;
  dst[i] = (_Float16)v;
}

// Wz (K=1024,N=1024): rows [0,768)=lstm_wih, [768,1024)=lstm_whh
__global__ void packB_wz(const float* __restrict__ wih, const float* __restrict__ whh,
                         _Float16* __restrict__ dst) {
  int i = blockIdx.x * 256 + threadIdx.x;
  if (i >= 1024 * 1024) return;
  int e = i & 15, lane = (i >> 4) & 31;
  int rest = i >> 9;
  int kt = rest & 31, nt = rest >> 5;      // KT=32
  int k = kt * 32 + e + ((lane >> 4) << 4);
  int j = nt * 16 + (lane & 15);
  float v = (k < 768) ? wih[k * 1024 + j] : whh[(k - 768) * 1024 + j];
  dst[i] = (_Float16)v;
}

__global__ void zero_state(float* __restrict__ c, _Float16* __restrict__ hf16, int n) {
  int i = blockIdx.x * 256 + threadIdx.x;
  if (i < n) { c[i] = 0.0f; hf16[i] = (_Float16)0.0f; }
}

// ---------------------------------------------------------------------------
// value = (enc_hidden @ value_w + value_b) * mask -> (B,H,T,HD) f32
// grid(B*120, 4): blockIdx.x -> (b, 16-row tile); 4 N-tiles per wave.
// ---------------------------------------------------------------------------
__global__ __launch_bounds__(32) void k0_value(
    const _Float16* __restrict__ enc16, const _Float16* __restrict__ vwp,
    const float* __restrict__ value_b, const float* __restrict__ mask,
    float* __restrict__ val) {
  int lane = threadIdx.x, hi = lane >> 4, lr = lane & 15;
  int b_ = blockIdx.x / 120;
  int t0 = (blockIdx.x % 120) * 16;
  int m0 = b_ * TT + t0;
  int nt0 = blockIdx.y * 4;
  const _Float16* arow = enc16 + (size_t)(m0 + lr) * DD;
  v8f acc[4] = {};
#pragma unroll
  for (int kt = 0; kt < 8; ++kt) {
    v16h a = load_a_rm(arow + kt * 32, hi);
#pragma unroll
    for (int t = 0; t < 4; ++t) {
      v16h b = load_b_pk(vwp, (nt0 + t) * 8 + kt, lane);
      acc[t] = WMMA(a, b, acc[t]);
    }
  }
#pragma unroll
  for (int t = 0; t < 4; ++t) {
    int col = (nt0 + t) * 16 + lr;
    int h = col >> 5, d = col & 31;
    float vb = value_b[col];
#pragma unroll
    for (int r = 0; r < 8; ++r) {
      int trow = t0 + r + 8 * hi;
      val[(((size_t)(b_ * HH + h) * TT) + trow) * HD + d] =
          (acc[t][r] + vb) * mask[b_ * TT + trow];
    }
  }
}

// ---------------------------------------------------------------------------
// Step 1: offawp = [h | query] @ Wcomb  (M=256,K=512,N=512); 4 N-tiles/wave.
// ---------------------------------------------------------------------------
__global__ __launch_bounds__(32) void k1_offaw(
    const _Float16* __restrict__ hf16, const _Float16* __restrict__ qf16,
    const _Float16* __restrict__ wcombp, float* __restrict__ offawp) {
  int lane = threadIdx.x, hi = lane >> 4, lr = lane & 15;
  int m0 = blockIdx.x * 16, nt0 = blockIdx.y * 4;
  int row = m0 + lr;
  v8f acc[4] = {};
#pragma unroll
  for (int kt = 0; kt < 16; ++kt) {
    const _Float16* ap = (kt < 8) ? (hf16 + (size_t)row * DD + kt * 32)
                                  : (qf16 + (size_t)row * DD + (kt - 8) * 32);
    v16h a = load_a_rm(ap, hi);
#pragma unroll
    for (int t = 0; t < 4; ++t) {
      v16h b = load_b_pk(wcombp, (nt0 + t) * 16 + kt, lane);
      acc[t] = WMMA(a, b, acc[t]);
    }
  }
#pragma unroll
  for (int t = 0; t < 4; ++t) {
    int col = (nt0 + t) * 16 + lr;
#pragma unroll
    for (int r = 0; r < 8; ++r)
      offawp[(m0 + r + 8 * hi) * 512 + col] = acc[t][r];
  }
}

// ---------------------------------------------------------------------------
// Step 2: deformable sampling + aw softmax -> hsampH (NE,H,16,HD) f16
// ---------------------------------------------------------------------------
__global__ __launch_bounds__(32) void k2_sample(
    const float* __restrict__ offawp, const float* __restrict__ off_b,
    const float* __restrict__ aw_b, const float* __restrict__ refpts,
    const float* __restrict__ vratio, const float* __restrict__ val,
    _Float16* __restrict__ hsampH) {
  int blk = blockIdx.x;
  int n = blk >> 3, h = blk & 7;
  int lane = threadIdx.x;
  int b = n >> 5;
  const float shapes[4] = {1024.f, 512.f, 256.f, 128.f};
  const int starts[4] = {0, 1024, 1536, 1792};
  float awv[16];
  float mx = -INFINITY;
#pragma unroll
  for (int j = 0; j < 16; ++j) {
    float a = offawp[n * 512 + 128 + h * 16 + j] + aw_b[h * 16 + j];
    awv[j] = a; mx = fmaxf(mx, a);
  }
  float s = 0.f;
#pragma unroll
  for (int j = 0; j < 16; ++j) { awv[j] = expf(awv[j] - mx); s += awv[j]; }
  float inv = 1.f / s;
  float refp = refpts[n];
#pragma unroll
  for (int l = 0; l < 4; ++l) {
    int Tl = (int)shapes[l];
    float rv = refp * vratio[b * 4 + l];
    const float* vbase = val + (((size_t)(b * HH + h) * TT) + starts[l]) * HD;
#pragma unroll
    for (int p = 0; p < 4; ++p) {
      int j = l * 4 + p;
      float off = offawp[n * 512 + h * 16 + j] + off_b[h * 16 + j];
      float x = (rv + off / shapes[l]) * shapes[l] - 0.5f;
      float x0 = floorf(x);
      float w = x - x0;
      int i0 = (int)x0;
      float g0 = 0.f, g1 = 0.f;
      if (i0 >= 0 && i0 < Tl)         g0 = vbase[i0 * HD + lane];
      if (i0 + 1 >= 0 && i0 + 1 < Tl) g1 = vbase[(i0 + 1) * HD + lane];
      hsampH[((((size_t)n * HH + h) * 16) + j) * HD + lane] =
          (_Float16)((g0 * (1.f - w) + g1 * w) * (awv[j] * inv));
    }
  }
}

// ---------------------------------------------------------------------------
// Step 3: fused attention per (n,h): WMMA 16x32 @ 32x256, tanh/alpha reduce,
// softmax over 16 points, weighted sum -> attn_res.
// ---------------------------------------------------------------------------
__global__ __launch_bounds__(32) void k3_attn(
    const _Float16* __restrict__ hsampH, const _Float16* __restrict__ ctxp,
    const float* __restrict__ ctx_b, const float* __restrict__ offawp,
    const float* __restrict__ hs_b, const float* __restrict__ alpha_w,
    const float* __restrict__ alpha_b, float* __restrict__ ares) {
  int blk = blockIdx.x;
  int n = blk >> 3, h = blk & 7;
  int lane = threadIdx.x, hi = lane >> 4, lr = lane & 15;
  const _Float16* hs = hsampH + (size_t)(n * HH + h) * 16 * HD;
  v16h a = load_a_rm(hs + lr * HD, hi);
  float ap[8];
#pragma unroll
  for (int r = 0; r < 8; ++r) ap[r] = 0.f;
#pragma unroll
  for (int jt = 0; jt < 16; ++jt) {
    int col = jt * 16 + lr;
    v16h b = load_b_pk(ctxp, jt, lane);
    v8f acc = {};
    acc = WMMA(a, b, acc);
    float hp  = offawp[n * 512 + 256 + col] + hs_b[col];
    float cb  = ctx_b[col];
    float awc = alpha_w[col];
#pragma unroll
    for (int r = 0; r < 8; ++r)
      ap[r] += tanhf(acc[r] + cb + hp) * awc;
  }
  float ab = alpha_b[0];
#pragma unroll
  for (int r = 0; r < 8; ++r) {
    float v = ap[r];
#pragma unroll
    for (int m = 1; m < 16; m <<= 1) v += __shfl_xor(v, m, 32);
    ap[r] = v + ab;
  }
  float av[16];
#pragma unroll
  for (int r = 0; r < 8; ++r) {
    av[r]     = __shfl(ap[r], 0, 32);
    av[r + 8] = __shfl(ap[r], 16, 32);
  }
  float mx = av[0];
#pragma unroll
  for (int j = 1; j < 16; ++j) mx = fmaxf(mx, av[j]);
  float s = 0.f;
#pragma unroll
  for (int j = 0; j < 16; ++j) { av[j] = expf(av[j] - mx); s += av[j]; }
  float inv = 1.f / s;
  float acc_d = 0.f;
#pragma unroll
  for (int j = 0; j < 16; ++j) acc_d += (av[j] * inv) * (float)hs[j * HD + lane];
  ares[n * DD + h * HD + lane] = acc_d;
}

// Step 3b: pack xz = [emb | attn_res | query | h] -> row-major f16 (NE x 1024)
__global__ __launch_bounds__(256) void k3b_packxz(
    const int* __restrict__ seq_tokens, int step, const _Float16* __restrict__ emb16,
    const float* __restrict__ ares, const _Float16* __restrict__ qf16,
    const _Float16* __restrict__ hf16, _Float16* __restrict__ xz16) {
  int idx = blockIdx.x * 256 + threadIdx.x;   // NE*1024
  int n = idx >> 10, k = idx & 1023;
  _Float16 v;
  if (k < 256) {
    int tok = (step == 0) ? 1 : seq_tokens[n * SS + step - 1];
    v = emb16[(size_t)tok * DD + k];
  } else if (k < 512) v = (_Float16)ares[n * DD + k - 256];
  else if (k < 768)   v = qf16[n * DD + k - 512];
  else                v = hf16[n * DD + k - 768];
  xz16[idx] = v;
}

// ---------------------------------------------------------------------------
// Step 4: gates = xz @ Wz  (M=256,K=1024,N=1024); 4 N-tiles/wave, grid(16,16)
// ---------------------------------------------------------------------------
__global__ __launch_bounds__(32) void k4_gates(
    const _Float16* __restrict__ xz16, const _Float16* __restrict__ wzp,
    float* __restrict__ gates) {
  int lane = threadIdx.x, hi = lane >> 4, lr = lane & 15;
  int m0 = blockIdx.x * 16, nt0 = blockIdx.y * 4;
  const _Float16* arow = xz16 + (size_t)(m0 + lr) * 1024;
  v8f acc[4] = {};
#pragma unroll
  for (int kt = 0; kt < 32; ++kt) {
    v16h a = load_a_rm(arow + kt * 32, hi);
#pragma unroll
    for (int t = 0; t < 4; ++t) {
      v16h b = load_b_pk(wzp, (nt0 + t) * 32 + kt, lane);
      acc[t] = WMMA(a, b, acc[t]);
    }
  }
#pragma unroll
  for (int t = 0; t < 4; ++t) {
    int col = (nt0 + t) * 16 + lr;
#pragma unroll
    for (int r = 0; r < 8; ++r)
      gates[(m0 + r + 8 * hi) * 1024 + col] = acc[t][r];
  }
}

// Step 4b: LSTM pointwise update (i,f,g,o)
__global__ __launch_bounds__(256) void k4b_lstm(
    const float* __restrict__ gates, float* __restrict__ c, _Float16* __restrict__ hf16) {
  int idx = blockIdx.x * 256 + threadIdx.x;   // NE*D
  int n = idx >> 8, d = idx & 255;
  const float* g = gates + (size_t)n * 1024;
  float gi = g[d], gf = g[256 + d], gg = g[512 + d], go = g[768 + d];
  float si = 1.f / (1.f + expf(-gi));
  float sf = 1.f / (1.f + expf(-gf));
  float so = 1.f / (1.f + expf(-go));
  float c2 = sf * c[idx] + si * tanhf(gg);
  float h2 = so * tanhf(c2);
  c[idx] = c2;
  hf16[idx] = (_Float16)h2;
}

// ---------------------------------------------------------------------------
// Step 5a: logits = h2 @ logit_w + logit_b  (M=256,K=256,N=10000)
// grid(125,16): 5 N-tiles per wave (625 = 125*5).
// ---------------------------------------------------------------------------
__global__ __launch_bounds__(32) void k5a_logits(
    const _Float16* __restrict__ hf16, const _Float16* __restrict__ lwp,
    const float* __restrict__ logit_b, float* __restrict__ logits) {
  int lane = threadIdx.x, hi = lane >> 4, lr = lane & 15;
  int nt0 = blockIdx.x * 5, m0 = blockIdx.y * 16;
  const _Float16* arow = hf16 + (size_t)(m0 + lr) * DD;
  // prefetch next block's B stream (gfx1250 global_prefetch)
  __builtin_prefetch(lwp + ((((size_t)((nt0 + 5) % 625) * 8) * 32 + lane) << 4), 0, 1);
  v8f acc[5] = {};
#pragma unroll
  for (int kt = 0; kt < 8; ++kt) {
    v16h a = load_a_rm(arow + kt * 32, hi);
#pragma unroll
    for (int t = 0; t < 5; ++t) {
      v16h b = load_b_pk(lwp, (nt0 + t) * 8 + kt, lane);
      acc[t] = WMMA(a, b, acc[t]);
    }
  }
#pragma unroll
  for (int t = 0; t < 5; ++t) {
    int col = (nt0 + t) * 16 + lr;
    float lb = logit_b[col];
#pragma unroll
    for (int r = 0; r < 8; ++r)
      logits[(size_t)(m0 + r + 8 * hi) * VV + col] = acc[t][r] + lb;
  }
}

// Step 5b: log_softmax row-wise, write output at (n, step, :)
__global__ __launch_bounds__(256) void k5b_logsm(
    const float* __restrict__ logits, float* __restrict__ out, int step) {
  __shared__ float red[256];
  int n = blockIdx.x, t = threadIdx.x;
  const float* row = logits + (size_t)n * VV;
  float mx = -INFINITY;
  for (int v = t; v < VV; v += 256) mx = fmaxf(mx, row[v]);
  red[t] = mx; __syncthreads();
  for (int s = 128; s > 0; s >>= 1) { if (t < s) red[t] = fmaxf(red[t], red[t + s]); __syncthreads(); }
  mx = red[0]; __syncthreads();
  float sum = 0.f;
  for (int v = t; v < VV; v += 256) sum += expf(row[v] - mx);
  red[t] = sum; __syncthreads();
  for (int s = 128; s > 0; s >>= 1) { if (t < s) red[t] += red[t + s]; __syncthreads(); }
  float logz = mx + logf(red[0]);
  float* o = out + ((size_t)n * STEPS + step) * VV;
  for (int v = t; v < VV; v += 256) o[v] = row[v] - logz;
}

// ---------------------------------------------------------------------------
// host launcher
// ---------------------------------------------------------------------------
extern "C" void kernel_launch(void* const* d_in, const int* in_sizes, int n_in,
                              void* d_out, int out_size, void* d_ws, size_t ws_size,
                              hipStream_t stream) {
  const int*   seq_tokens = (const int*)d_in[0];
  const float* query      = (const float*)d_in[1];
  const float* refpts     = (const float*)d_in[2];
  const float* enc_hidden = (const float*)d_in[3];
  const float* mask       = (const float*)d_in[4];
  const float* vratio     = (const float*)d_in[5];
  const float* embed_w    = (const float*)d_in[6];
  const float* logit_w    = (const float*)d_in[7];
  const float* logit_b    = (const float*)d_in[8];
  const float* value_w    = (const float*)d_in[9];
  const float* value_b    = (const float*)d_in[10];
  const float* off_w      = (const float*)d_in[11];
  const float* off_b      = (const float*)d_in[12];
  const float* aw_w       = (const float*)d_in[13];
  const float* aw_b       = (const float*)d_in[14];
  const float* ctx_w      = (const float*)d_in[15];
  const float* ctx_b      = (const float*)d_in[16];
  const float* hs_w       = (const float*)d_in[17];
  const float* hs_b       = (const float*)d_in[18];
  const float* alpha_w    = (const float*)d_in[19];
  const float* alpha_b    = (const float*)d_in[20];
  const float* lstm_wih   = (const float*)d_in[21];
  const float* lstm_whh   = (const float*)d_in[22];
  float* out = (float*)d_out;

  char* ws = (char*)d_ws;
  size_t off = 0;
  auto alloc = [&](size_t bytes) -> void* {
    void* p = ws + off;
    off = (off + bytes + 255) & ~(size_t)255;
    return p;
  };
  _Float16* hf16   = (_Float16*)alloc((size_t)NE * DD * 2);
  _Float16* qf16   = (_Float16*)alloc((size_t)NE * DD * 2);
  float*    cstate = (float*)   alloc((size_t)NE * DD * 4);
  _Float16* enc16  = (_Float16*)alloc((size_t)BB * TT * DD * 2);
  _Float16* vwp    = (_Float16*)alloc((size_t)DD * DD * 2);          // packed value_w
  float*    val    = (float*)   alloc((size_t)BB * HH * TT * HD * 4);
  _Float16* wcombp = (_Float16*)alloc((size_t)512 * 512 * 2);        // packed
  _Float16* wzp    = (_Float16*)alloc((size_t)1024 * 1024 * 2);      // packed
  _Float16* emb16  = (_Float16*)alloc((size_t)VV * DD * 2);
  _Float16* lwp    = (_Float16*)alloc((size_t)DD * VV * 2);          // packed
  _Float16* ctxp   = (_Float16*)alloc((size_t)HD * DD * 2);          // packed
  float*    offawp = (float*)   alloc((size_t)NE * 512 * 4);
  _Float16* hsampH = (_Float16*)alloc((size_t)NE * HH * 16 * HD * 2);
  float*    ares   = (float*)   alloc((size_t)NE * DD * 4);
  _Float16* xz16   = (_Float16*)alloc((size_t)NE * 1024 * 2);
  float*    gates  = (float*)   alloc((size_t)NE * 1024 * 4);
  float*    logits = (float*)   alloc((size_t)NE * VV * 4);
  (void)ws_size; (void)n_in; (void)in_sizes; (void)out_size;

  auto blocks = [](size_t n) { return (unsigned)((n + 255) / 256); };

  // ---- setup ----
  pack16<<<blocks((size_t)NE * DD), 256, 0, stream>>>(query, qf16, NE * DD);
  pack16<<<blocks((size_t)BB * TT * DD), 256, 0, stream>>>(enc_hidden, enc16, BB * TT * DD);
  pack16<<<blocks((size_t)VV * DD), 256, 0, stream>>>(embed_w, emb16, VV * DD);
  packB_f32<<<blocks((size_t)DD * DD), 256, 0, stream>>>(value_w, vwp, DD, DD);
  packB_f32<<<blocks((size_t)DD * VV), 256, 0, stream>>>(logit_w, lwp, DD, VV);
  packB_f32<<<blocks((size_t)HD * DD), 256, 0, stream>>>(ctx_w, ctxp, HD, DD);
  packB_wcomb<<<blocks((size_t)512 * 512), 256, 0, stream>>>(off_w, aw_w, hs_w, wcombp);
  packB_wz<<<blocks((size_t)1024 * 1024), 256, 0, stream>>>(lstm_wih, lstm_whh, wzp);
  zero_state<<<blocks((size_t)NE * DD), 256, 0, stream>>>(cstate, hf16, NE * DD);

  // value tensor (B,H,T,HD)
  k0_value<<<dim3(BB * 120, 4), 32, 0, stream>>>(enc16, vwp, value_b, mask, val);

  // ---- sequential scan over 20 steps ----
  for (int step = 0; step < STEPS; ++step) {
    k1_offaw<<<dim3(NE / 16, 8), 32, 0, stream>>>(hf16, qf16, wcombp, offawp);
    k2_sample<<<NE * HH, 32, 0, stream>>>(offawp, off_b, aw_b, refpts, vratio, val, hsampH);
    k3_attn<<<NE * HH, 32, 0, stream>>>(hsampH, ctxp, ctx_b, offawp, hs_b, alpha_w, alpha_b, ares);
    k3b_packxz<<<blocks((size_t)NE * 1024), 256, 0, stream>>>(seq_tokens, step, emb16, ares,
                                                              qf16, hf16, xz16);
    k4_gates<<<dim3(NE / 16, 16), 32, 0, stream>>>(xz16, wzp, gates);
    k4b_lstm<<<blocks((size_t)NE * DD), 256, 0, stream>>>(gates, cstate, hf16);
    k5a_logits<<<dim3(125, NE / 16), 32, 0, stream>>>(hf16, lwp, logit_b, logits);
    k5b_logsm<<<NE, 256, 0, stream>>>(logits, out, step);
  }
}
